// MultiHeadAttention_37160057045218
// MI455X (gfx1250) — matrix operand order repaired and checked by reading, compiled, not run on hardware
//
#include <hip/hip_runtime.h>
#include <hip/hip_bf16.h>

typedef __attribute__((ext_vector_type(16))) __bf16 v16bf;
typedef __attribute__((ext_vector_type(8)))  __bf16 v8bf;
typedef __attribute__((ext_vector_type(8)))  float  v8f;

#define D_MODEL 1024
#define N_HEADS 16
#define D_K 64
#define T_SEQ 2048

__device__ __forceinline__ __bf16 f2bf(float f) { return (__bf16)f; }

__device__ __forceinline__ v8f wmma_bf16(v16bf a, v16bf b, v8f c) {
  return __builtin_amdgcn_wmma_f32_16x16x32_bf16(false, a, false, b, (short)0, c,
                                                 false, false);
}

// 16 contiguous bf16 -> fragment (two ds_load_b128)
__device__ __forceinline__ v16bf ld16(const __bf16* p) {
  v16bf r;
  __builtin_memcpy(&r, p, 32);
  return r;
}
// two contiguous 8-element chunks -> fragment
__device__ __forceinline__ v16bf ld8x2(const __bf16* p0, const __bf16* p1) {
  v16bf r;
  __builtin_memcpy(&r, p0, 16);
  __builtin_memcpy((char*)&r + 16, p1, 16);
  return r;
}

// ---- CDNA5 async global->LDS copy (ASYNCcnt path, ISA 08 §4) ----
__device__ __forceinline__ unsigned to_lds_off(const void* p) {
  return (unsigned)(size_t)(const __attribute__((address_space(3))) void*)p;
}
__device__ __forceinline__ void async_ld_b128(void* lds_dst, const void* gsrc) {
  unsigned l = to_lds_off(lds_dst);
  asm volatile("global_load_async_to_lds_b128 %0, %1, off"
               :: "v"(l), "v"(gsrc) : "memory");
}
__device__ __forceinline__ void wait_async0() {
  asm volatile("s_wait_asynccnt 0" ::: "memory");
}

// ---------------------------------------------------------------------------
// One-time transpose+convert of a weight matrix: WT[n][k] = (bf16)W[k][n].
// ---------------------------------------------------------------------------
__global__ void __launch_bounds__(128)
wt_kernel(const float* __restrict__ W, __bf16* __restrict__ WT, int K, int N) {
  __shared__ alignas(16) __bf16 Ts[64][72];
  const int tid = threadIdx.x;
  const int k0 = blockIdx.y * 64, n0 = blockIdx.x * 64;

  const int r = tid >> 1, cs = (tid & 1) * 32;
#pragma unroll
  for (int p = 0; p < 4; ++p) {
    v8bf t;
#pragma unroll
    for (int j = 0; j < 8; ++j)
      t[j] = f2bf(W[(size_t)(k0 + r) * N + n0 + cs + p * 8 + j]);
    *(v8bf*)&Ts[r][cs + p * 8] = t;
  }
  __syncthreads();
  const int c = tid & 63, rs = (tid >> 6) * 32;
#pragma unroll
  for (int p = 0; p < 4; ++p) {
    v8bf o;
#pragma unroll
    for (int j = 0; j < 8; ++j) o[j] = Ts[rs + p * 8 + j][c];
    *(v8bf*)&WT[(size_t)(n0 + c) * K + k0 + rs + p * 8] = o;
  }
}

// ---------------------------------------------------------------------------
// Per-head transpose of Q: QT[(bh*64+dk)*T + t] = Qbf[(n*T+t)*1024 + h*64 + dk]
// ---------------------------------------------------------------------------
__global__ void __launch_bounds__(128)
qt_kernel(const __bf16* __restrict__ Qbf, __bf16* __restrict__ QT, int T) {
  __shared__ alignas(16) __bf16 Ts[64][72];
  const int tid = threadIdx.x;
  const int t0 = blockIdx.x * 64;
  const int bh = blockIdx.y;
  const size_t base =
      ((size_t)(bh / N_HEADS) * T) * (N_HEADS * D_K) + (size_t)(bh % N_HEADS) * D_K;

  const int r = tid >> 1, cs = (tid & 1) * 32;
#pragma unroll
  for (int p = 0; p < 4; ++p)
    *(v8bf*)&Ts[r][cs + p * 8] =
        *(const v8bf*)&Qbf[base + (size_t)(t0 + r) * (N_HEADS * D_K) + cs + p * 8];
  __syncthreads();
  const int c = tid & 63, rs = (tid >> 6) * 32;
#pragma unroll
  for (int p = 0; p < 4; ++p) {
    v8bf o;
#pragma unroll
    for (int j = 0; j < 8; ++j) o[j] = Ts[rs + p * 8 + j][c];
    *(v8bf*)&QT[((size_t)bh * 64 + c) * T + t0 + rs + p * 8] = o;
  }
}

// ---------------------------------------------------------------------------
// Tiled GEMM, double-buffered; B (and bf16 A) staged with async->LDS copies.
// C[M x N] = A[M x K] * W + bias, W pre-transposed bf16 (WT[n][k]).
// Tile 64x64, K-step 64, 8 waves, 4 WMMA / wave / step.
// ---------------------------------------------------------------------------
template <bool A_BF16, bool OUT_BF16>
__global__ void __launch_bounds__(256)
gemm_tiled(const void* __restrict__ A_, const __bf16* __restrict__ WT,
           const float* __restrict__ bias, void* __restrict__ C_,
           int M, int K, int N) {
  __shared__ alignas(16) __bf16 As[2][64][72];
  __shared__ alignas(16) __bf16 BsT[2][64][72];

  const int tid  = threadIdx.x;
  const int lane = tid & 31;
  const int wave = tid >> 5;
  const int wrow = wave & 3;
  const int wcol = wave >> 2;
  const int hf   = lane >> 4;
  const int nl   = lane & 15;

  const int mBase = blockIdx.y * 64;
  const int nBase = blockIdx.x * 64;

  const float*  Af = (const float*)A_;
  const __bf16* Ab = (const __bf16*)A_;

  const int ar = tid >> 2, ac = (tid & 3) * 16;    // A staging: row, 16-col chunk
  const int bc = tid & 63, brs = (tid >> 6) * 16;  // B staging: n-col, 16-k chunk

  v8f acc0 = {}; v8f acc1 = {};

  // ---- stage tile 0 ----
  if (A_BF16) {
    const __bf16* src = Ab + (size_t)(mBase + ar) * K + ac;
    async_ld_b128(&As[0][ar][ac],     src);
    async_ld_b128(&As[0][ar][ac + 8], src + 8);
  } else {
    const float* src = Af + (size_t)(mBase + ar) * K + ac;
    v8bf p0, p1;
#pragma unroll
    for (int j = 0; j < 8; ++j) { p0[j] = f2bf(src[j]); p1[j] = f2bf(src[8 + j]); }
    *(v8bf*)&As[0][ar][ac] = p0;
    *(v8bf*)&As[0][ar][ac + 8] = p1;
  }
  {
    const __bf16* src = WT + (size_t)(nBase + bc) * K + brs;
    async_ld_b128(&BsT[0][bc][brs],     src);
    async_ld_b128(&BsT[0][bc][brs + 8], src + 8);
  }
  wait_async0();
  __syncthreads();

  const int nIter = K >> 6;
  for (int it = 0; it < nIter; ++it) {
    const int k0  = it << 6;
    const int cur = it & 1;
    const int nb  = cur ^ 1;
    const bool nxt = (it + 1 < nIter);

    // ---- issue staging for tile it+1 ----
    float afl[16];
    if (nxt) {
      if (A_BF16) {
        const __bf16* src = Ab + (size_t)(mBase + ar) * K + k0 + 64 + ac;
        async_ld_b128(&As[nb][ar][ac],     src);
        async_ld_b128(&As[nb][ar][ac + 8], src + 8);
      } else {
        const float* src = Af + (size_t)(mBase + ar) * K + k0 + 64 + ac;
        __builtin_memcpy(afl, src, 64);
      }
      const __bf16* src = WT + (size_t)(nBase + bc) * K + k0 + 64 + brs;
      async_ld_b128(&BsT[nb][bc][brs],     src);
      async_ld_b128(&BsT[nb][bc][brs + 8], src + 8);
      if (it + 2 < nIter) {
        __builtin_prefetch(WT + (size_t)(nBase + bc) * K + k0 + 128 + brs, 0, 3);
        if (A_BF16) __builtin_prefetch(Ab + (size_t)(mBase + ar) * K + k0 + 128 + ac, 0, 3);
        else        __builtin_prefetch(Af + (size_t)(mBase + ar) * K + k0 + 128 + ac, 0, 3);
      }
    }

    // ---- compute on tile it (async copies for it+1 in flight) ----
    {
      const __bf16* arow = &As[cur][wrow * 16 + nl][0];
      v16bf a0 = ld16(arow + hf * 16);
      v16bf a1 = ld16(arow + 32 + hf * 16);
      const __bf16* br0 = &BsT[cur][wcol * 32 + nl][0];
      const __bf16* br1 = &BsT[cur][wcol * 32 + 16 + nl][0];
      v16bf b00 = ld16(br0 + hf * 16);
      v16bf b01 = ld16(br1 + hf * 16);
      v16bf b10 = ld16(br0 + 32 + hf * 16);
      v16bf b11 = ld16(br1 + 32 + hf * 16);
      acc0 = wmma_bf16(a0, b00, acc0);
      acc1 = wmma_bf16(a0, b01, acc1);
      acc0 = wmma_bf16(a1, b10, acc0);
      acc1 = wmma_bf16(a1, b11, acc1);
    }

    // ---- fp32-A path: convert + commit through registers ----
    if (!A_BF16 && nxt) {
      v8bf p0, p1;
#pragma unroll
      for (int j = 0; j < 8; ++j) { p0[j] = f2bf(afl[j]); p1[j] = f2bf(afl[8 + j]); }
      *(v8bf*)&As[nb][ar][ac]     = p0;
      *(v8bf*)&As[nb][ar][ac + 8] = p1;
    }
    wait_async0();
    __syncthreads();
  }

#pragma unroll
  for (int nt = 0; nt < 2; ++nt) {
    const v8f acc = nt ? acc1 : acc0;
    const int col = nBase + wcol * 32 + nt * 16 + nl;
    const float bv = bias[col];
#pragma unroll
    for (int r = 0; r < 8; ++r) {
      const int row = mBase + wrow * 16 + hf * 8 + r;
      const float v = acc[r] + bv;
      if (OUT_BF16) ((__bf16*)C_)[(size_t)row * N + col] = f2bf(v);
      else          ((float*)C_)[(size_t)row * N + col]  = v;
    }
  }
}

// ---------------------------------------------------------------------------
// Flash attention (v = q per reference). Block = 4 waves, per (64 q-rows, n*h).
// Double-buffered K/V tiles fed entirely by async global->LDS copies.
// ---------------------------------------------------------------------------
__global__ void __launch_bounds__(128)
attn_kernel(const __bf16* __restrict__ Qbf, const __bf16* __restrict__ Kbf,
            const __bf16* __restrict__ QT, const int* __restrict__ amask,
            const int* __restrict__ dec_ptr, __bf16* __restrict__ Abf, int T) {
  __shared__ alignas(16) __bf16 Qs[64][72];       // [qrow][dk]
  __shared__ alignas(16) __bf16 Ks[2][64][72];    // [key][dk]
  __shared__ alignas(16) __bf16 VsT[2][64][72];   // [dk][key]
  __shared__ alignas(16) __bf16 Ps[4][16][72];    // per-wave P [row][key]

  const int tid  = threadIdx.x;
  const int lane = tid & 31;
  const int wave = tid >> 5;
  const int hf   = lane >> 4;
  const int nl   = lane & 15;

  const int qBase  = blockIdx.x * 64;
  const int bh     = blockIdx.y;
  const int nIdx   = bh / N_HEADS;
  const int h      = bh % N_HEADS;
  const int encLen = T - dec_ptr[0];

  const size_t rowStride = (size_t)(N_HEADS * D_K);
  const size_t base = ((size_t)nIdx * T) * rowStride + (size_t)h * D_K;

  const int sr = tid >> 1, scs = (tid & 1) * 32;   // row-major staging map
  const int vc = tid & 63, vrs = (tid >> 6) * 32;  // transposed staging map

  // ---- stage Q tile + K/V tile 0 (async copies) ----
#pragma unroll
  for (int p = 0; p < 4; ++p) {
    async_ld_b128(&Qs[sr][scs + p * 8],
                  &Qbf[base + (size_t)(qBase + sr) * rowStride + scs + p * 8]);
    async_ld_b128(&Ks[0][sr][scs + p * 8],
                  &Kbf[base + (size_t)sr * rowStride + scs + p * 8]);
    async_ld_b128(&VsT[0][vc][vrs + p * 8],
                  &QT[((size_t)bh * 64 + vc) * T + vrs + p * 8]);
  }
  wait_async0();
  __syncthreads();

  const __bf16* qrow = &Qs[wave * 16 + nl][0];
  v16bf qa0 = ld16(qrow + hf * 16);
  v16bf qa1 = ld16(qrow + 32 + hf * 16);

  v8f o[4] = {};
  float mrow[8], lrow[8];
#pragma unroll
  for (int r = 0; r < 8; ++r) { mrow[r] = -1e30f; lrow[r] = 0.f; }
  const int qi0 = qBase + wave * 16 + hf * 8;

  const int nTiles = T >> 6;
  for (int it = 0; it < nTiles; ++it) {
    const int kt  = it << 6;
    const int cur = it & 1;
    const int nb  = cur ^ 1;
    const bool nxt = (it + 1 < nTiles);

    // ---- async staging for tile it+1 (no VGPR round-trip) ----
    if (nxt) {
      const size_t kb = base + (size_t)(kt + 64 + sr) * rowStride + scs;
      const size_t vb = ((size_t)bh * 64 + vc) * T + kt + 64 + vrs;
#pragma unroll
      for (int p = 0; p < 4; ++p) {
        async_ld_b128(&Ks[nb][sr][scs + p * 8],  &Kbf[kb + p * 8]);
        async_ld_b128(&VsT[nb][vc][vrs + p * 8], &QT[vb + p * 8]);
      }
      if (it + 2 < nTiles) {
        __builtin_prefetch(&Kbf[kb + 64 * rowStride], 0, 3);
        __builtin_prefetch(&QT[vb + 64], 0, 3);
      }
    }

    // ---- S = Q * K^T ----
    v8f s[4] = {};
#pragma unroll
    for (int ntp = 0; ntp < 4; ntp += 2) {
      v16bf b0 = ld16(&Ks[cur][(ntp + 0) * 16 + nl][hf * 16]);
      v16bf b1 = ld16(&Ks[cur][(ntp + 0) * 16 + nl][32 + hf * 16]);
      v16bf b2 = ld16(&Ks[cur][(ntp + 1) * 16 + nl][hf * 16]);
      v16bf b3 = ld16(&Ks[cur][(ntp + 1) * 16 + nl][32 + hf * 16]);
      s[ntp + 0] = wmma_bf16(qa0, b0, s[ntp + 0]);
      s[ntp + 0] = wmma_bf16(qa1, b1, s[ntp + 0]);
      s[ntp + 1] = wmma_bf16(qa0, b2, s[ntp + 1]);
      s[ntp + 1] = wmma_bf16(qa1, b3, s[ntp + 1]);
    }

    // ---- scale + masks + tile row-max ----
    float mt[8];
#pragma unroll
    for (int r = 0; r < 8; ++r) mt[r] = -1e30f;
#pragma unroll
    for (int nt = 0; nt < 4; ++nt) {
      const int kj = kt + nt * 16 + nl;
      const bool colok = (amask[nIdx * T + kj] != 0);
#pragma unroll
      for (int r = 0; r < 8; ++r) {
        float v = s[nt][r] * 0.125f;
        const bool vis = colok && ((kj < encLen) || (kj <= qi0 + r));
        v = vis ? v : -1e30f;
        s[nt][r] = v;
        mt[r] = fmaxf(mt[r], v);
      }
    }
#pragma unroll
    for (int off = 1; off < 16; off <<= 1)
#pragma unroll
      for (int r = 0; r < 8; ++r)
        mt[r] = fmaxf(mt[r], __shfl_xor(mt[r], off, 32));

    float rs[8];
#pragma unroll
    for (int r = 0; r < 8; ++r) {
      const float mnew  = fmaxf(mrow[r], mt[r]);
      const float alpha = __expf(mrow[r] - mnew);
      mrow[r] = mnew;
      lrow[r] *= alpha;
      o[0][r] *= alpha; o[1][r] *= alpha; o[2][r] *= alpha; o[3][r] *= alpha;
      rs[r] = 0.f;
    }
#pragma unroll
    for (int nt = 0; nt < 4; ++nt)
#pragma unroll
      for (int r = 0; r < 8; ++r) {
        const float p = __expf(s[nt][r] - mrow[r]);
        rs[r] += p;
        Ps[wave][hf * 8 + r][nt * 16 + nl] = f2bf(p);
      }
#pragma unroll
    for (int off = 1; off < 16; off <<= 1)
#pragma unroll
      for (int r = 0; r < 8; ++r)
        rs[r] += __shfl_xor(rs[r], off, 32);
#pragma unroll
    for (int r = 0; r < 8; ++r) lrow[r] += rs[r];

    // ---- O += P * V ----
    const __bf16* prow = &Ps[wave][nl][0];
    v16bf pa0 = ld16(prow + hf * 16);
    v16bf pa1 = ld16(prow + 32 + hf * 16);
#pragma unroll
    for (int ntp = 0; ntp < 4; ntp += 2) {
      v16bf b0 = ld16(&VsT[cur][(ntp + 0) * 16 + nl][hf * 16]);
      v16bf b1 = ld16(&VsT[cur][(ntp + 0) * 16 + nl][32 + hf * 16]);
      v16bf b2 = ld16(&VsT[cur][(ntp + 1) * 16 + nl][hf * 16]);
      v16bf b3 = ld16(&VsT[cur][(ntp + 1) * 16 + nl][32 + hf * 16]);
      o[ntp + 0] = wmma_bf16(pa0, b0, o[ntp + 0]);
      o[ntp + 0] = wmma_bf16(pa1, b1, o[ntp + 0]);
      o[ntp + 1] = wmma_bf16(pa0, b2, o[ntp + 1]);
      o[ntp + 1] = wmma_bf16(pa1, b3, o[ntp + 1]);
    }

    wait_async0();
    __syncthreads();
  }

  // ---- normalize and write ----
#pragma unroll
  for (int nt = 0; nt < 4; ++nt)
#pragma unroll
    for (int r = 0; r < 8; ++r) {
      const float denom = lrow[r];
      const float v = (denom > 0.f) ? (o[nt][r] / denom) : 0.f;
      const size_t row = (size_t)nIdx * T + qBase + wave * 16 + hf * 8 + r;
      Abf[row * rowStride + h * D_K + nt * 16 + nl] = f2bf(v);
    }
}

// ---------------------------------------------------------------------------
extern "C" void kernel_launch(void* const* d_in, const int* in_sizes, int n_in,
                              void* d_out, int out_size, void* d_ws, size_t ws_size,
                              hipStream_t stream) {
  const float* x   = (const float*)d_in[0];
  const float* Wq  = (const float*)d_in[1];
  const float* bq  = (const float*)d_in[2];
  const float* Wk  = (const float*)d_in[3];
  const float* bk  = (const float*)d_in[4];
  const float* Wo  = (const float*)d_in[5];
  const float* bo  = (const float*)d_in[6];
  const int* amask = (const int*)d_in[7];
  const int* dec   = (const int*)d_in[8];

  const int T  = T_SEQ;
  const int NT = in_sizes[7];    // N * T
  const int Nb = NT / T;
  const int Kd = D_MODEL;
  const int Nc = N_HEADS * D_K;  // 1024

  __bf16* Qbf = (__bf16*)d_ws;              // NT x 1024
  __bf16* Kbf = Qbf + (size_t)NT * Nc;      // NT x 1024
  __bf16* Abf = Kbf + (size_t)NT * Nc;      // NT x 1024
  __bf16* QT  = Abf + (size_t)NT * Nc;      // (Nb*H*64) x T
  __bf16* WTq = QT  + (size_t)NT * Nc;      // 1024 x 1024
  __bf16* WTk = WTq + (size_t)Kd * Nc;
  __bf16* WTo = WTk + (size_t)Kd * Nc;

  {
    dim3 b(128), g(Nc / 64, Kd / 64);
    wt_kernel<<<g, b, 0, stream>>>(Wq, WTq, Kd, Nc);
    wt_kernel<<<g, b, 0, stream>>>(Wk, WTk, Kd, Nc);
    wt_kernel<<<g, b, 0, stream>>>(Wo, WTo, Kd, Nc);
  }
  {
    dim3 b(256), g(Nc / 64, NT / 64);
    gemm_tiled<false, true><<<g, b, 0, stream>>>(x, WTq, bq, Qbf, NT, Kd, Nc);
    gemm_tiled<false, true><<<g, b, 0, stream>>>(x, WTk, bk, Kbf, NT, Kd, Nc);
  }
  {
    dim3 b(128), g(T / 64, Nb * N_HEADS);
    qt_kernel<<<g, b, 0, stream>>>(Qbf, QT, T);
  }
  {
    dim3 b(128), g(T / 64, Nb * N_HEADS);
    attn_kernel<<<g, b, 0, stream>>>(Qbf, Kbf, QT, amask, dec, Abf, T);
  }
  {
    dim3 b(256), g(D_MODEL / 64, NT / 64);
    gemm_tiled<true, false><<<g, b, 0, stream>>>(Abf, WTo, bo, (float*)d_out,
                                                 NT, D_MODEL, D_MODEL);
  }
}
